// Model_16758962389521
// MI455X (gfx1250) — matrix operand order repaired and checked by reading, not compiled, not run on hardware
//
#include <hip/hip_runtime.h>
#include <hip/hip_bf16.h>

// ---------------------------------------------------------------------------
// Types for CDNA5 WMMA (wave32, 16x16x32 f16 -> f32)
// ---------------------------------------------------------------------------
typedef _Float16 h16;
typedef __attribute__((ext_vector_type(16))) _Float16 v16h;
typedef __attribute__((ext_vector_type(8)))  _Float16 v8h;
typedef __attribute__((ext_vector_type(8)))  float    v8f;

#define WMMA16(a, b, c) \
  __builtin_amdgcn_wmma_f32_16x16x32_f16(false, (a), false, (b), (short)0, (c), false, false)

// A fragment (16x32 f16, M x K): lane holds row = lane%16,
// k-chunk base = (lane/16)*8; elements [k0..k0+7] and [k0+16..k0+23].
__device__ __forceinline__ v16h ld_frag_a(const h16* base, int stride, int row, int k0) {
  const h16* p = base + (long)row * stride + k0;
  v8h lo = *(const v8h*)(p);
  v8h hi = *(const v8h*)(p + 16);
  return __builtin_shufflevector(lo, hi, 0,1,2,3,4,5,6,7,8,9,10,11,12,13,14,15);
}

// B fragment (32x16 f16, K x N) from an (N,K) row-major weight (i.e. B = Wt):
// lane holds col = n0 + lane%16, k-chunk base kb = (lane/16)*8;
// elements [kb..kb+7] and [kb+16..kb+23].
__device__ __forceinline__ v16h ld_frag_b(const h16* w, int stride, int col, int kb) {
  const h16* p = w + (long)col * stride + kb;
  v8h lo = *(const v8h*)(p);
  v8h hi = *(const v8h*)(p + 16);
  return __builtin_shufflevector(lo, hi, 0,1,2,3,4,5,6,7,8,9,10,11,12,13,14,15);
}

// ---------------------------------------------------------------------------
// Generic WMMA GEMM:  C[M,N] = A[M,K] (f16, row-major) * W[N,K]^T (f16)
// block = 256 threads = 8 waves, block tile 64(M) x 32(N); wave tile 16x16.
// Optional bias, relu, f32 and/or f16 output, optional (l,b)->(b,l) row remap.
// M%16==0, N%16==0, K%32==0 guaranteed by caller.
// ---------------------------------------------------------------------------
__global__ void gemm_wmma(const h16* __restrict__ A, const h16* __restrict__ W,
                          const float* __restrict__ bias,
                          float* __restrict__ outF, h16* __restrict__ outH,
                          int M, int N, int K, int relu, int remapB, int remapL) {
  int lane = threadIdx.x & 31, wave = threadIdx.x >> 5;
  int mw = wave & 3, nw = wave >> 2;
  int m0 = blockIdx.x * 64 + mw * 16;
  int n0 = blockIdx.y * 32 + nw * 16;
  if (m0 >= M || n0 >= N) return;

  int row = m0 + (lane & 15);
  int col = n0 + (lane & 15);
  int ka  = (lane >> 4) * 8;
  int kb  = (lane >> 4) * 8;

  v8f c = {};
  for (int kt = 0; kt < K; kt += 32) {
    v16h a = ld_frag_a(A, K, row, kt + ka);
    v16h b = ld_frag_b(W, K, col, kt + kb);
    c = WMMA16(a, b, c);
  }

  int rbase = m0 + (lane >> 4) * 8;
  float bv = bias ? bias[col] : 0.f;
  for (int v = 0; v < 8; ++v) {
    int r = rbase + v;
    float val = c[v] + bv;
    if (relu && val < 0.f) val = 0.f;
    long idx;
    if (remapB > 0) {
      int b_ = r % remapB, l_ = r / remapB;           // row = l*B + b
      idx = ((long)b_ * remapL + l_) * N + col;       // out (b, l, n)
    } else {
      idx = (long)r * N + col;
    }
    if (outF) outF[idx] = val;
    if (outH) outH[idx] = (h16)val;
  }
}

// ---------------------------------------------------------------------------
// Persistent single-workgroup GRU over T steps (B=8), WMMA per step.
// h kept in LDS (f16, rows 0..7 valid, 8..15 zero). Whh:(3H,H) Wih:(3H,Din)
// f16 weights streamed from L2 each step. Gates in f32.  out: (T,B,ostride) f16.
// Dynamic LDS: ah 16*H, ax 16*Din, gh 8*3H, gx 8*3H  (all f16).
// ---------------------------------------------------------------------------
__global__ void gru_seq(const h16* __restrict__ Whh, const h16* __restrict__ Wih,
                        const float* __restrict__ bih, const float* __restrict__ bhh,
                        const h16* __restrict__ xseq, h16* __restrict__ out,
                        int T, int H, int Din, int dir, int ostride, int ooff) {
  extern __shared__ __align__(16) char smem[];
  h16* ah = (h16*)smem;              // 16*H
  h16* ax = ah + 16 * H;             // 16*Din
  h16* gh = ax + 16 * Din;           // 8*3H
  h16* gx = gh + 8 * 3 * H;          // 8*3H
  const int B = 8;
  const int N3 = 3 * H;
  int tid = threadIdx.x, lane = tid & 31, wave = tid >> 5;

  for (int i = tid; i < 16 * H;   i += 256) ah[i] = (h16)0.f;
  for (int i = tid; i < 16 * Din; i += 256) ax[i] = (h16)0.f;
  __syncthreads();

  int ntiles = N3 / 16;
  int rA = lane & 15;
  int ka = (lane >> 4) * 8;
  int kb = (lane >> 4) * 8;
  int cl = lane & 15;

  for (int t = 0; t < T; ++t) {
    int tr = dir ? (T - 1 - t) : t;
    const h16* xt = xseq + (long)tr * B * Din;
    for (int i = tid; i < B * Din; i += 256) ax[i] = xt[i];
    __syncthreads();

    for (int tb = wave; tb < ntiles; tb += 8) {   // wave-uniform
      int n0 = tb * 16;
      v8f ch = {};
      for (int kt = 0; kt < H; kt += 32)
        ch = WMMA16(ld_frag_a(ah, H, rA, kt + ka),
                    ld_frag_b(Whh, H, n0 + cl, kt + kb), ch);
      v8f cx = {};
      for (int kt = 0; kt < Din; kt += 32)
        cx = WMMA16(ld_frag_a(ax, Din, rA, kt + ka),
                    ld_frag_b(Wih, Din, n0 + cl, kt + kb), cx);
      if (lane < 16) {  // lanes 0..15 hold batch rows 0..7 (v = batch)
        for (int v = 0; v < 8; ++v) {
          gh[v * N3 + n0 + cl] = (h16)ch[v];
          gx[v * N3 + n0 + cl] = (h16)cx[v];
        }
      }
    }
    __syncthreads();

    for (int i = tid; i < B * H; i += 256) {
      int b = i / H, j = i % H;
      float xr = (float)gx[b * N3 + j]         + bih[j];
      float hr = (float)gh[b * N3 + j]         + bhh[j];
      float xz = (float)gx[b * N3 + H + j]     + bih[H + j];
      float hz = (float)gh[b * N3 + H + j]     + bhh[H + j];
      float xn = (float)gx[b * N3 + 2*H + j]   + bih[2*H + j];
      float hn = (float)gh[b * N3 + 2*H + j]   + bhh[2*H + j];
      float r = 1.f / (1.f + __expf(-(xr + hr)));
      float z = 1.f / (1.f + __expf(-(xz + hz)));
      float n = tanhf(xn + r * hn);
      float hp = (float)ah[b * H + j];
      float hv = (1.f - z) * n + z * hp;
      ah[b * H + j] = (h16)hv;
      out[((long)tr * B + b) * ostride + ooff + j] = (h16)hv;
    }
    __syncthreads();
  }
}

// ---------------------------------------------------------------------------
// Small helper kernels
// ---------------------------------------------------------------------------
__global__ void cast_f2h(const float* __restrict__ s, h16* __restrict__ d, long n) {
  long i = (long)blockIdx.x * blockDim.x + threadIdx.x;
  if (i < n) d[i] = (h16)s[i];
}
__global__ void cast_pad_f2h(const float* __restrict__ s, h16* __restrict__ d,
                             int N, int K, int Kp) {
  long i = (long)blockIdx.x * blockDim.x + threadIdx.x;
  if (i >= (long)N * Kp) return;
  int nn = i / Kp, kp = i % Kp;
  d[i] = (kp < K) ? (h16)s[(long)nn * K + kp] : (h16)0.f;
}
__global__ void zero_f32(float* p, int n) {
  int i = blockIdx.x * blockDim.x + threadIdx.x;
  if (i < n) p[i] = 0.f;
}
// im2col for 1D conv, kernel 5, pad 2.  act: (B*T, Cin) f32 -> col: (B*T, Kp) f16
__global__ void im2col_k5(const float* __restrict__ act, h16* __restrict__ col,
                          int Bb, int T, int Cin, int Kp) {
  long i = (long)blockIdx.x * blockDim.x + threadIdx.x;
  long n = (long)Bb * T * Kp;
  if (i >= n) return;
  int m = i / Kp, kp = i % Kp;
  h16 v = (h16)0.f;
  if (kp < Cin * 5) {
    int c = kp / 5, kk = kp % 5;
    int b = m / T, t = m % T, tt = t + kk - 2;
    if (tt >= 0 && tt < T) v = (h16)act[((long)b * T + tt) * Cin + c];
  }
  col[i] = v;
}
// training-mode BN (mean/var over M rows) + ReLU, per channel block.
__global__ void bn_relu_kernel(float* __restrict__ act, const float* __restrict__ g,
                               const float* __restrict__ b, int M, int C) {
  int c = blockIdx.x;
  __shared__ float red[256];
  float s = 0.f, s2 = 0.f;
  for (int m = threadIdx.x; m < M; m += 256) {
    float v = act[(long)m * C + c];
    s += v; s2 += v * v;
  }
  red[threadIdx.x] = s; __syncthreads();
  for (int o = 128; o > 0; o >>= 1) { if (threadIdx.x < o) red[threadIdx.x] += red[threadIdx.x + o]; __syncthreads(); }
  float mean = red[0] / M; __syncthreads();
  red[threadIdx.x] = s2; __syncthreads();
  for (int o = 128; o > 0; o >>= 1) { if (threadIdx.x < o) red[threadIdx.x] += red[threadIdx.x + o]; __syncthreads(); }
  float var = red[0] / M - mean * mean;
  float sc = g[c] * rsqrtf(var + 1e-5f), sh = b[c];
  for (int m = threadIdx.x; m < M; m += 256) {
    float v = (act[(long)m * C + c] - mean) * sc + sh;
    act[(long)m * C + c] = v > 0.f ? v : 0.f;
  }
}

// ---------------------------------------------------------------------------
// VQ-EMA pipeline
// ---------------------------------------------------------------------------
__global__ void vq_wsum(const float* __restrict__ W, float* __restrict__ wsum) {
  int k = blockIdx.x * blockDim.x + threadIdx.x;
  if (k >= 512) return;
  float s = 0.f;
  for (int d = 0; d < 64; ++d) { float v = W[k * 64 + d]; s += v * v; }
  wsum[k] = s;
}
__device__ __forceinline__ unsigned mixhash(unsigned x) {
  x ^= x >> 16; x *= 0x7feb352du; x ^= x >> 15; x *= 0x846ca68bu; x ^= x >> 16;
  return x;
}
// Gumbel-max categorical over logits(-dists); row-constant |z|^2 drops out.
__global__ void vq_sample(const float* __restrict__ P, const float* __restrict__ wsum,
                          int* __restrict__ samples) {
  int id = blockIdx.x * blockDim.x + threadIdx.x;
  if (id >= 5 * 320) return;
  int s = id / 320, m = id % 320;
  float best = -1e30f; int bi = 0;
  for (int k = 0; k < 512; ++k) {
    float logit = 2.f * P[(long)m * 512 + k] - wsum[k];
    unsigned h = mixhash(0x9e3779b9u * (unsigned)s ^ 0x85ebca6bu * (unsigned)m ^
                         0xc2b2ae35u * (unsigned)k ^ 42u);
    float u = (float)(h >> 8) * (1.f / 16777216.f) + 1e-7f;
    float gmb = -__logf(-__logf(u));
    float sc = logit + gmb;
    if (sc > best) { best = sc; bi = k; }
  }
  samples[id] = bi;
}
__global__ void vq_count(const int* __restrict__ samples, float* __restrict__ cnt) {
  int id = blockIdx.x * blockDim.x + threadIdx.x;
  if (id < 5 * 320) atomicAdd(&cnt[samples[id]], 0.2f);
}
__global__ void vq_emacnt(const float* __restrict__ cnt, float* __restrict__ emacnt) {
  __shared__ float red[512];
  int k = threadIdx.x;
  float ec = 0.001f * cnt[k];
  red[k] = ec; __syncthreads();
  for (int o = 256; o > 0; o >>= 1) { if (k < o) red[k] += red[k + o]; __syncthreads(); }
  float n = red[0];
  emacnt[k] = (ec + 1e-5f) / (n + 512.f * 1e-5f) * n;
}
__global__ void vq_accw(const int* __restrict__ samples, const float* __restrict__ z,
                        float* __restrict__ accW) {
  int id = blockIdx.x * blockDim.x + threadIdx.x;
  if (id >= 5 * 320 * 64) return;
  int sm = id / 64, d = id % 64;
  int m = sm % 320;
  atomicAdd(&accW[(long)samples[sm] * 64 + d], 0.2f * z[(long)m * 64 + d]);
}
__global__ void vq_wnew(const float* __restrict__ W, const float* __restrict__ accW,
                        const float* __restrict__ emacnt, float* __restrict__ Wnew) {
  int i = blockIdx.x * blockDim.x + threadIdx.x;
  if (i >= 512 * 64) return;
  Wnew[i] = (0.999f * W[i] + 0.001f * accW[i]) / emacnt[i / 64];
}
__global__ void vq_quant_loss(const int* __restrict__ samples, const float* __restrict__ Wnew,
                              const float* __restrict__ z, float* __restrict__ quant,
                              float* __restrict__ lossacc) {
  int i = blockIdx.x * blockDim.x + threadIdx.x;
  if (i >= 320 * 64) return;
  int m = i / 64, d = i % 64;
  float q = 0.f;
  for (int s = 0; s < 5; ++s) q += Wnew[(long)samples[s * 320 + m] * 64 + d];
  q *= 0.2f;
  quant[i] = q;
  float df = z[i] - q;
  atomicAdd(lossacc, df * df);
}
__global__ void vq_final(const float* __restrict__ cnt, const float* __restrict__ lossacc,
                         float* __restrict__ out_loss, float* __restrict__ out_perp) {
  __shared__ float red[512];
  int k = threadIdx.x;
  float avg = cnt[k] / 320.f;
  red[k] = avg * __logf(avg + 1e-10f);
  __syncthreads();
  for (int o = 256; o > 0; o >>= 1) { if (k < o) red[k] += red[k + o]; __syncthreads(); }
  if (k == 0) {
    *out_perp = __expf(-red[0]);
    *out_loss = 0.25f * lossacc[0] / (320.f * 64.f);
  }
}

// ---------------------------------------------------------------------------
// Sequence builders
// ---------------------------------------------------------------------------
// xseq0 (T=40,B=8,128) f16: [0:64]=quant[b*T+t], [64:128]=spk[speakers[b]]
__global__ void build_xseq0(const float* __restrict__ quant, const float* __restrict__ spk,
                            const int* __restrict__ speakers, h16* __restrict__ xseq) {
  int i = blockIdx.x * blockDim.x + threadIdx.x;
  if (i >= 40 * 8 * 128) return;
  int t = i / (8 * 128), r = i % (8 * 128), b = r / 128, j = r % 128;
  float v = (j < 64) ? quant[((long)b * 40 + t) * 64 + j]
                     : spk[(long)speakers[b] * 64 + (j - 64)];
  xseq[i] = (h16)v;
}
// seq2 (L=6400,B=8,512) f16: [0:256]=aud_emb[x[b,l]], [256:512]=s1[(pad+l/200)][b]
__global__ void build_seq2(const int* __restrict__ x, const float* __restrict__ aud_emb,
                           const h16* __restrict__ s1, h16* __restrict__ seq2) {
  long i = (long)blockIdx.x * blockDim.x + threadIdx.x;
  if (i >= (long)6400 * 8 * 512) return;
  int t = (int)(i / 4096), r = (int)(i % 4096), b = r / 512, j = r % 512;
  if (j < 256) {
    seq2[i] = (h16)aud_emb[(long)x[(long)b * 6400 + t] * 256 + j];
  } else {
    seq2[i] = s1[((long)(4 + t / 200) * 8 + b) * 256 + (j - 256)];
  }
}

// ---------------------------------------------------------------------------
// Host launcher
// ---------------------------------------------------------------------------
static inline size_t bump(size_t& off, size_t bytes) {
  size_t r = off; off += (bytes + 255) & ~(size_t)255; return r;
}
static inline dim3 g1d(long n, int b) { return dim3((unsigned)((n + b - 1) / b)); }

extern "C" void kernel_launch(void* const* d_in, const int* in_sizes, int n_in,
                              void* d_out, int out_size, void* d_ws, size_t ws_size,
                              hipStream_t stream) {
  (void)in_sizes; (void)n_in; (void)ws_size;
  const int*   x        = (const int*)d_in[0];
  const float* mels     = (const float*)d_in[1];
  const int*   speakers = (const int*)d_in[2];
  const float* cw1 = (const float*)d_in[3];
  const float* g1  = (const float*)d_in[4];  const float* b1 = (const float*)d_in[5];
  const float* cw2 = (const float*)d_in[6];
  const float* g2  = (const float*)d_in[7];  const float* b2 = (const float*)d_in[8];
  const float* cw3 = (const float*)d_in[9];
  const float* g3  = (const float*)d_in[10]; const float* b3 = (const float*)d_in[11];
  const float* cw4 = (const float*)d_in[12]; const float* cb4 = (const float*)d_in[13];
  const float* codebook = (const float*)d_in[14];
  const float* spk = (const float*)d_in[15];
  // GRU params (wih, whh, bih, bhh) x5
  const float* r_wih[5]; const float* r_whh[5]; const float* r_bih[5]; const float* r_bhh[5];
  for (int i = 0; i < 5; ++i) {
    r_wih[i] = (const float*)d_in[16 + 4 * i + 0];
    r_whh[i] = (const float*)d_in[16 + 4 * i + 1];
    r_bih[i] = (const float*)d_in[16 + 4 * i + 2];
    r_bhh[i] = (const float*)d_in[16 + 4 * i + 3];
  }
  const float* aud_emb = (const float*)d_in[36];
  const float* fc1w = (const float*)d_in[37]; const float* fc1b = (const float*)d_in[38];
  const float* fc2w = (const float*)d_in[39]; const float* fc2b = (const float*)d_in[40];
  float* out = (float*)d_out;

  const int B = 8, T = 40, L = 6400;
  char* ws = (char*)d_ws;
  size_t off = 0;
  // f16 weight copies
  h16* wc1h = (h16*)(ws + bump(off, (size_t)512 * 416 * 2));   // padded K 400->416
  h16* wc2h = (h16*)(ws + bump(off, (size_t)512 * 2560 * 2));
  h16* wc3h = (h16*)(ws + bump(off, (size_t)512 * 2560 * 2));
  h16* wc4h = (h16*)(ws + bump(off, (size_t)64 * 512 * 2));
  h16* cbh  = (h16*)(ws + bump(off, (size_t)512 * 64 * 2));
  const int gru_H[5]   = {128, 128, 128, 128, 896};
  const int gru_Din[5] = {128, 128, 256, 256, 512};
  h16* g_wih[5]; h16* g_whh[5];
  for (int i = 0; i < 5; ++i) {
    g_wih[i] = (h16*)(ws + bump(off, (size_t)3 * gru_H[i] * gru_Din[i] * 2));
    g_whh[i] = (h16*)(ws + bump(off, (size_t)3 * gru_H[i] * gru_H[i] * 2));
  }
  h16* wfc1h = (h16*)(ws + bump(off, (size_t)256 * 896 * 2));
  h16* wfc2h = (h16*)(ws + bump(off, (size_t)256 * 256 * 2));
  // activations
  h16*   col   = (h16*)(ws + bump(off, (size_t)320 * 2560 * 2));
  float* hA    = (float*)(ws + bump(off, (size_t)320 * 512 * 4));
  float* hB    = (float*)(ws + bump(off, (size_t)320 * 512 * 4));
  h16*   acth  = (h16*)(ws + bump(off, (size_t)320 * 512 * 2));
  float* zbuf  = (float*)(ws + bump(off, (size_t)320 * 64 * 4));
  h16*   zh    = (h16*)(ws + bump(off, (size_t)320 * 64 * 2));
  float* Pbuf  = (float*)(ws + bump(off, (size_t)320 * 512 * 4));
  float* wsum  = (float*)(ws + bump(off, (size_t)512 * 4));
  int*   smp   = (int*)(ws + bump(off, (size_t)5 * 320 * 4));
  float* cnt   = (float*)(ws + bump(off, (size_t)512 * 4));
  float* emac  = (float*)(ws + bump(off, (size_t)512 * 4));
  float* accW  = (float*)(ws + bump(off, (size_t)512 * 64 * 4));
  float* Wnew  = (float*)(ws + bump(off, (size_t)512 * 64 * 4));
  float* quant = (float*)(ws + bump(off, (size_t)320 * 64 * 4));
  float* lossa = (float*)(ws + bump(off, (size_t)256));
  h16* xseq0 = (h16*)(ws + bump(off, (size_t)T * B * 128 * 2));
  h16* s0out = (h16*)(ws + bump(off, (size_t)T * B * 256 * 2));
  h16* s1out = (h16*)(ws + bump(off, (size_t)T * B * 256 * 2));
  h16* seq2h = (h16*)(ws + bump(off, (size_t)L * B * 512 * 2));
  h16* hs2   = (h16*)(ws + bump(off, (size_t)L * B * 896 * 2));
  h16* fc1h  = (h16*)(ws + bump(off, (size_t)(L * B) * 256 * 2));

  // ---- weight casts to f16 ----
  cast_pad_f2h<<<g1d((long)512 * 416, 256), 256, 0, stream>>>(cw1, wc1h, 512, 400, 416);
  cast_f2h<<<g1d((long)512 * 2560, 256), 256, 0, stream>>>(cw2, wc2h, (long)512 * 2560);
  cast_f2h<<<g1d((long)512 * 2560, 256), 256, 0, stream>>>(cw3, wc3h, (long)512 * 2560);
  cast_f2h<<<g1d((long)64 * 512, 256), 256, 0, stream>>>(cw4, wc4h, (long)64 * 512);
  cast_f2h<<<g1d((long)512 * 64, 256), 256, 0, stream>>>(codebook, cbh, (long)512 * 64);
  for (int i = 0; i < 5; ++i) {
    long nih = (long)3 * gru_H[i] * gru_Din[i];
    long nhh = (long)3 * gru_H[i] * gru_H[i];
    cast_f2h<<<g1d(nih, 256), 256, 0, stream>>>(r_wih[i], g_wih[i], nih);
    cast_f2h<<<g1d(nhh, 256), 256, 0, stream>>>(r_whh[i], g_whh[i], nhh);
  }
  cast_f2h<<<g1d((long)256 * 896, 256), 256, 0, stream>>>(fc1w, wfc1h, (long)256 * 896);
  cast_f2h<<<g1d((long)256 * 256, 256), 256, 0, stream>>>(fc2w, wfc2h, (long)256 * 256);

  // ---- encoder: conv/BN/ReLU x3 + 1x1 conv ----
  // conv1: mels is already (B*T, 80)
  im2col_k5<<<g1d((long)320 * 416, 256), 256, 0, stream>>>(mels, col, B, T, 80, 416);
  gemm_wmma<<<dim3(5, 16), 256, 0, stream>>>(col, wc1h, nullptr, hA, nullptr, 320, 512, 416, 0, 0, 0);
  bn_relu_kernel<<<512, 256, 0, stream>>>(hA, g1, b1, 320, 512);
  im2col_k5<<<g1d((long)320 * 2560, 256), 256, 0, stream>>>(hA, col, B, T, 512, 2560);
  gemm_wmma<<<dim3(5, 16), 256, 0, stream>>>(col, wc2h, nullptr, hB, nullptr, 320, 512, 2560, 0, 0, 0);
  bn_relu_kernel<<<512, 256, 0, stream>>>(hB, g2, b2, 320, 512);
  im2col_k5<<<g1d((long)320 * 2560, 256), 256, 0, stream>>>(hB, col, B, T, 512, 2560);
  gemm_wmma<<<dim3(5, 16), 256, 0, stream>>>(col, wc3h, nullptr, hA, nullptr, 320, 512, 2560, 0, 0, 0);
  bn_relu_kernel<<<512, 256, 0, stream>>>(hA, g3, b3, 320, 512);
  cast_f2h<<<g1d((long)320 * 512, 256), 256, 0, stream>>>(hA, acth, (long)320 * 512);
  gemm_wmma<<<dim3(5, 2), 256, 0, stream>>>(acth, wc4h, cb4, zbuf, nullptr, 320, 64, 512, 0, 0, 0);

  // ---- VQ-EMA ----
  cast_f2h<<<g1d((long)320 * 64, 256), 256, 0, stream>>>(zbuf, zh, (long)320 * 64);
  vq_wsum<<<2, 256, 0, stream>>>(codebook, wsum);
  gemm_wmma<<<dim3(5, 16), 256, 0, stream>>>(zh, cbh, nullptr, Pbuf, nullptr, 320, 512, 64, 0, 0, 0);
  zero_f32<<<g1d(512, 256), 256, 0, stream>>>(cnt, 512);
  zero_f32<<<g1d(512 * 64, 256), 256, 0, stream>>>(accW, 512 * 64);
  zero_f32<<<1, 256, 0, stream>>>(lossa, 1);
  vq_sample<<<g1d(1600, 256), 256, 0, stream>>>(Pbuf, wsum, smp);
  vq_count<<<g1d(1600, 256), 256, 0, stream>>>(smp, cnt);
  vq_emacnt<<<1, 512, 0, stream>>>(cnt, emac);
  vq_accw<<<g1d((long)5 * 320 * 64, 256), 256, 0, stream>>>(smp, zbuf, accW);
  vq_wnew<<<g1d(512 * 64, 256), 256, 0, stream>>>(codebook, accW, emac, Wnew);
  vq_quant_loss<<<g1d(320 * 64, 256), 256, 0, stream>>>(smp, Wnew, zbuf, quant, lossa);
  vq_final<<<1, 512, 0, stream>>>(cnt, lossa, out + (out_size - 2), out + (out_size - 1));

  // ---- vocoder GRUs ----
  build_xseq0<<<g1d((long)T * B * 128, 256), 256, 0, stream>>>(quant, spk, speakers, xseq0);
  // rnn1 layer0 (Din=128,H=128): fwd -> s0out[:,0:128], bwd -> s0out[:,128:256]
  {
    size_t sm = (size_t)(16 * 128 + 16 * 128 + 2 * 8 * 384) * 2;
    gru_seq<<<1, 256, sm, stream>>>(g_whh[0], g_wih[0], r_bih[0], r_bhh[0], xseq0, s0out,
                                    T, 128, 128, 0, 256, 0);
    gru_seq<<<1, 256, sm, stream>>>(g_whh[1], g_wih[1], r_bih[1], r_bhh[1], xseq0, s0out,
                                    T, 128, 128, 1, 256, 128);
  }
  // rnn1 layer1 (Din=256,H=128)
  {
    size_t sm = (size_t)(16 * 128 + 16 * 256 + 2 * 8 * 384) * 2;
    gru_seq<<<1, 256, sm, stream>>>(g_whh[2], g_wih[2], r_bih[2], r_bhh[2], s0out, s1out,
                                    T, 128, 256, 0, 256, 0);
    gru_seq<<<1, 256, sm, stream>>>(g_whh[3], g_wih[3], r_bih[3], r_bhh[3], s0out, s1out,
                                    T, 128, 256, 1, 256, 128);
  }
  // rnn2 input sequence + persistent big GRU (H=896, Din=512, 6400 steps)
  build_seq2<<<g1d((long)L * B * 512, 256), 256, 0, stream>>>(x, aud_emb, s1out, seq2h);
  {
    size_t sm = (size_t)(16 * 896 + 16 * 512 + 2 * 8 * 2688) * 2;  // 128 KB LDS
    gru_seq<<<1, 256, sm, stream>>>(g_whh[4], g_wih[4], r_bih[4], r_bhh[4], seq2h, hs2,
                                    L, 896, 512, 0, 896, 0);
  }

  // ---- FC head: relu(hs @ fc1w^T + b) @ fc2w^T + b, scatter (l,b)->(b,l) ----
  gemm_wmma<<<dim3(800, 8), 256, 0, stream>>>(hs2, wfc1h, fc1b, nullptr, fc1h,
                                              L * B, 256, 896, 1, 0, 0);
  gemm_wmma<<<dim3(800, 8), 256, 0, stream>>>(fc1h, wfc2h, fc2b, out, nullptr,
                                              L * B, 256, 256, 0, B, L);
}
